// ModelMergerTome_81484119540336
// MI455X (gfx1250) — compile-verified, matrix operand order checked by reading, not run on hardware
//
#include <hip/hip_runtime.h>
#include <hip/hip_bf16.h>
#include <math.h>

// ---------------- model constants ----------------
#define MM 7
#define LL 12
#define BB 8
#define CC 768
#define HH 12
#define HD 64
#define RR 8
#define PP 16
#define IMGSZ 224
#define GRD 14
#define NP0 197
#define LRA 8
#define ADHN 64
#define NCLS 100
#define C3 (3*CC)
#define C4 (4*CC)
#define ATT_SCALE 0.125f
#define LNEPS 1e-6f

typedef __attribute__((ext_vector_type(16))) _Float16 v16h;
typedef __attribute__((ext_vector_type(8)))  _Float16 v8h;
typedef __attribute__((ext_vector_type(4)))  _Float16 v4h;
typedef __attribute__((ext_vector_type(8)))  float    v8f;
typedef __attribute__((ext_vector_type(4)))  float    v4f;
typedef unsigned int v4u __attribute__((ext_vector_type(4)));
typedef int          v8i __attribute__((ext_vector_type(8)));
typedef int          v4i __attribute__((ext_vector_type(4)));

#if __has_builtin(__builtin_amdgcn_tensor_load_to_lds) && __has_builtin(__builtin_amdgcn_s_wait_tensorcnt)
#define USE_TDM 1
#endif

// ---------------- generic strided-batched WMMA GEMM ----------------
// D = act(alpha * A(+affine) x B + bias) + add1 + add2
// Batch index bi decomposes as o = bi/div_, in = bi%div_; every operand has
// (inner, outer) element strides so attention (m*B+b, h) batching works.
struct GemmP {
  int Mr, Nc, K, div_;
  const float* A; int lda; long Ai, Ao;
  const float* Bm; int ldb; int tB; long Bi, Bo;
  const float* aw; const float* ab; long AWi, AWo;   // per-K affine on A
  const float* bias; long BsI, BsO;                  // per-col bias
  const float* a1; int ld1; long A1i, A1o;           // residual adds
  const float* a2; int ld2; long A2i, A2o;
  float* D; int ldc; long Di, Do;
  float alpha; int act;                              // 0 none, 1 gelu, 2 relu
};

#define TM 64
#define TN 64
#define TK 64
#define LDP 80   // LDS row pitch in halves: 160B, multiple of 16B for b128 reads

static __device__ __forceinline__ bool aligned16(const void* p) {
  return (((unsigned long)p) & 15UL) == 0;
}

__global__ __launch_bounds__(256) void gemm_wmma_kernel(GemmP p) {
  __shared__ _Float16 As[TM][LDP];
  __shared__ _Float16 Bs[TN][LDP];   // stored [n][k] so fragments are K-contiguous
  const int bi = blockIdx.z;
  const long o = bi / p.div_, in = bi % p.div_;
  const float* A    = p.A  + o*p.Ao  + in*p.Ai;
  const float* Bm   = p.Bm + o*p.Bo  + in*p.Bi;
  const float* aw   = p.aw   ? p.aw   + o*p.AWo + in*p.AWi : nullptr;
  const float* ab   = p.ab   ? p.ab   + o*p.AWo + in*p.AWi : nullptr;
  const float* bias = p.bias ? p.bias + o*p.BsO + in*p.BsI : nullptr;
  const float* a1   = p.a1   ? p.a1   + o*p.A1o + in*p.A1i : nullptr;
  const float* a2   = p.a2   ? p.a2   + o*p.A2o + in*p.A2i : nullptr;
  float* D = p.D + o*p.Do + in*p.Di;

  const int row0 = blockIdx.x * TM, col0 = blockIdx.y * TN;
  const int tid = threadIdx.x, lane = tid & 31, wv = tid >> 5;
  // 8 waves: 4 along M (16 rows each), 2 along N (32 cols each)
  const int wm = (wv & 3) * 16, wn = (wv >> 2) * 32;
  const int hl = lane & 15, sel = lane >> 4;

  const bool fullM = (row0 + TM <= p.Mr);
  const bool fullN = (col0 + TN <= p.Nc);
  const bool alignA = aligned16(A)  && ((p.lda & 3) == 0);
  const bool alignB = aligned16(Bm) && ((p.ldb & 3) == 0);
  const bool alignW = (aw == nullptr) || (aligned16(aw) && aligned16(ab));

  v8f acc[2] = {v8f{}, v8f{}};

  for (int kk = 0; kk < p.K; kk += TK) {
    const bool fullK = (kk + TK <= p.K);
    // ---------------- stage A tile (64 rows x 64 K, fused LN affine) -------
    if (fullM && fullK) {
      if (alignA && alignW) {
        if (aw) {
          #pragma unroll
          for (int t = 0; t < 4; ++t) {             // 1024 vec4 slots / 256 thr
            int idx = tid + t*256;
            int r = idx >> 4, c = (idx & 15) << 2;
            const float* src = A + (long)(row0 + r)*p.lda + kk + c;
            v4f v = *(const v4f*)src;
            v4f w = *(const v4f*)(aw + kk + c);
            v4f b = *(const v4f*)(ab + kk + c);
            if (kk + TK < p.K) __builtin_prefetch(src + TK, 0, 1);
            v4h h;
            #pragma unroll
            for (int e = 0; e < 4; ++e) h[e] = (_Float16)(v[e]*w[e] + b[e]);
            *(v4h*)&As[r][c] = h;
          }
        } else {
          #pragma unroll
          for (int t = 0; t < 4; ++t) {
            int idx = tid + t*256;
            int r = idx >> 4, c = (idx & 15) << 2;
            const float* src = A + (long)(row0 + r)*p.lda + kk + c;
            v4f v = *(const v4f*)src;
            if (kk + TK < p.K) __builtin_prefetch(src + TK, 0, 1);
            v4h h;
            #pragma unroll
            for (int e = 0; e < 4; ++e) h[e] = (_Float16)v[e];
            *(v4h*)&As[r][c] = h;
          }
        }
      } else {
        #pragma unroll
        for (int t = 0; t < 16; ++t) {              // 4096 scalar slots
          int idx = tid + t*256;
          int r = idx >> 6, c = idx & 63;
          float v = A[(long)(row0 + r)*p.lda + kk + c];
          if (aw) v = v * aw[kk + c] + ab[kk + c];
          As[r][c] = (_Float16)v;
        }
      }
    } else {
      for (int t = 0; t < 16; ++t) {
        int idx = tid + t*256;
        int r = idx >> 6, c = idx & 63;
        int gr = row0 + r, gk = kk + c;
        float v = 0.f;
        if (gr < p.Mr && gk < p.K) {
          v = A[(long)gr * p.lda + gk];
          if (aw) v = v * aw[gk] + ab[gk];
        }
        As[r][c] = (_Float16)v;
      }
    }
    // ---------------- stage B tile into [n][k] -----------------------------
    if (p.tB) {           // B[n*ldb + k] : K-contiguous in memory and in LDS
      if (fullN && fullK) {
        if (alignB) {
          #pragma unroll
          for (int t = 0; t < 4; ++t) {
            int idx = tid + t*256;
            int n = idx >> 4, c = (idx & 15) << 2;
            const float* src = Bm + (long)(col0 + n)*p.ldb + kk + c;
            v4f v = *(const v4f*)src;
            if (kk + TK < p.K) __builtin_prefetch(src + TK, 0, 1);
            v4h h;
            #pragma unroll
            for (int e = 0; e < 4; ++e) h[e] = (_Float16)v[e];
            *(v4h*)&Bs[n][c] = h;
          }
        } else {
          #pragma unroll
          for (int t = 0; t < 16; ++t) {
            int idx = tid + t*256;
            int n = idx >> 6, c = idx & 63;
            Bs[n][c] = (_Float16)Bm[(long)(col0 + n)*p.ldb + kk + c];
          }
        }
      } else {
        for (int t = 0; t < 16; ++t) {
          int idx = tid + t*256;
          int n = idx >> 6, c = idx & 63;
          int gn = col0 + n, gk = kk + c;
          float v = 0.f;
          if (gn < p.Nc && gk < p.K) v = Bm[(long)gn * p.ldb + gk];
          Bs[n][c] = (_Float16)v;
        }
      }
    } else {              // B[k*ldb + n] : N-contiguous -> transpose into LDS
      if (fullN && fullK) {
        if (alignB) {
          #pragma unroll
          for (int t = 0; t < 4; ++t) {
            int idx = tid + t*256;
            int kr = idx >> 4, n = (idx & 15) << 2;
            const float* src = Bm + (long)(kk + kr)*p.ldb + col0 + n;
            v4f v = *(const v4f*)src;
            if (kk + TK < p.K) __builtin_prefetch(src + (long)TK*p.ldb, 0, 1);
            #pragma unroll
            for (int e = 0; e < 4; ++e) Bs[n + e][kr] = (_Float16)v[e];
          }
        } else {
          #pragma unroll
          for (int t = 0; t < 16; ++t) {
            int idx = tid + t*256;
            int n = idx & 63, kr = idx >> 6;
            Bs[n][kr] = (_Float16)Bm[(long)(kk + kr)*p.ldb + col0 + n];
          }
        }
      } else {
        for (int t = 0; t < 16; ++t) {
          int idx = tid + t*256;
          int n = idx & 63, kr = idx >> 6;
          int gn = col0 + n, gk = kk + kr;
          float v = 0.f;
          if (gn < p.Nc && gk < p.K) v = Bm[(long)gk * p.ldb + gn];
          Bs[n][kr] = (_Float16)v;
        }
      }
    }
    __syncthreads();
    // ---------------- 4 WMMAs per wave per K-tile --------------------------
    // CDNA5 16-bit fragment layout: lanes<16 hold K kb..kb+7 & kb+16..kb+23,
    // lanes>=16 hold kb+8.. (kb = sel*8); A row = wm+hl, B col = wn(+16)+hl.
    #pragma unroll
    for (int s = 0; s < 2; ++s) {
      const int kb = s*32 + sel*8;
      v8h alo = *(const v8h*)&As[wm + hl][kb];
      v8h ahi = *(const v8h*)&As[wm + hl][kb + 16];
      v16h av;
      #pragma unroll
      for (int e = 0; e < 8; ++e) { av[e] = alo[e]; av[e+8] = ahi[e]; }
      v8h b0lo = *(const v8h*)&Bs[wn + hl][kb];
      v8h b0hi = *(const v8h*)&Bs[wn + hl][kb + 16];
      v8h b1lo = *(const v8h*)&Bs[wn + 16 + hl][kb];
      v8h b1hi = *(const v8h*)&Bs[wn + 16 + hl][kb + 16];
      v16h bv0, bv1;
      #pragma unroll
      for (int e = 0; e < 8; ++e) {
        bv0[e] = b0lo[e]; bv0[e+8] = b0hi[e];
        bv1[e] = b1lo[e]; bv1[e+8] = b1hi[e];
      }
      acc[0] = __builtin_amdgcn_wmma_f32_16x16x32_f16(false, av, false, bv0,
                                                      (short)0, acc[0], false, false);
      acc[1] = __builtin_amdgcn_wmma_f32_16x16x32_f16(false, av, false, bv1,
                                                      (short)0, acc[1], false, false);
    }
    __syncthreads();
  }
  // ---------------- epilogue ----------------
  #pragma unroll
  for (int f = 0; f < 2; ++f) {
    const int ccol = col0 + wn + f*16 + hl;
    if (ccol < p.Nc) {
      #pragma unroll
      for (int j = 0; j < 8; ++j) {
        int r = row0 + wm + sel*8 + j;
        if (r < p.Mr) {
          float v = acc[f][j] * p.alpha;
          if (bias) v += bias[ccol];
          if (p.act == 1)      v = 0.5f * v * (1.0f + erff(v * 0.70710678118f));
          else if (p.act == 2) v = fmaxf(v, 0.f);
          if (a1) v += a1[(long)r * p.ld1 + ccol];
          if (a2) v += a2[(long)r * p.ld2 + ccol];
          D[(long)r * p.ldc + ccol] = v;
        }
      }
    }
  }
}

// ------- QKV GEMM with Tensor-Data-Mover B staging (f16 weights) -----------
// Bh layout: [m][Nc][K] f16, K-contiguous. K must be a multiple of TK and
// A must be 16B aligned with lda%4==0 (holds for A = xn, lda = CC).
struct GemmHT {
  const float* A; int lda; int Mr;
  const _Float16* Bh; long Bho;
  int Nc, K;
  const float* aw; const float* ab; long AWo;
  const float* bias; long BsO;
  float* D; int ldc; long Do;
};

__global__ __launch_bounds__(256) void gemm_wmma_tdm_kernel(GemmHT p) {
  __shared__ _Float16 As[TM][LDP];
  __shared__ _Float16 Bs[TN][LDP];
  const int m = blockIdx.z;
  const float* A       = p.A;
  const _Float16* Bh   = p.Bh + (long)m * p.Bho;
  const float* aw      = p.aw + (long)m * p.AWo;
  const float* ab      = p.ab + (long)m * p.AWo;
  const float* bias    = p.bias + (long)m * p.BsO;
  float* D             = p.D + (long)m * p.Do;
  const int row0 = blockIdx.x * TM, col0 = blockIdx.y * TN;
  const int tid = threadIdx.x, lane = tid & 31, wv = tid >> 5;
  const int wm = (wv & 3) * 16, wn = (wv >> 2) * 32;
  const int hl = lane & 15, sel = lane >> 4;
  const bool fullM = (row0 + TM <= p.Mr);

  v8f acc[2] = {v8f{}, v8f{}};
  for (int kk = 0; kk < p.K; kk += TK) {
    // A staging with fused LN affine (vec4 fast path / guarded edge path)
    if (fullM) {
      #pragma unroll
      for (int t = 0; t < 4; ++t) {
        int idx = tid + t*256;
        int r = idx >> 4, c = (idx & 15) << 2;
        const float* src = A + (long)(row0 + r)*p.lda + kk + c;
        v4f v = *(const v4f*)src;
        v4f w = *(const v4f*)(aw + kk + c);
        v4f b = *(const v4f*)(ab + kk + c);
        if (kk + TK < p.K) __builtin_prefetch(src + TK, 0, 1);
        v4h h;
        #pragma unroll
        for (int e = 0; e < 4; ++e) h[e] = (_Float16)(v[e]*w[e] + b[e]);
        *(v4h*)&As[r][c] = h;
      }
    } else {
      for (int t = 0; t < 16; ++t) {
        int idx = tid + t*256;
        int r = idx >> 6, c = idx & 63;
        int gr = row0 + r;
        float v = 0.f;
        if (gr < p.Mr) v = A[(long)gr*p.lda + kk + c] * aw[kk + c] + ab[kk + c];
        As[r][c] = (_Float16)v;
      }
    }
#ifdef USE_TDM
    // B staging via TDM: DMA a 64(n) x 64(k) f16 tile into LDS, padding each
    // 128B row (interval code 4) with 32B (amount code 7) -> pitch 160B = LDP.
    if (wv == 0) {
      unsigned ldsoff = (unsigned)(unsigned long)(const void*)&Bs[0][0];
      unsigned long ga = (unsigned long)(const void*)(Bh + (long)col0*p.K + kk);
      unsigned td0 = (unsigned)(p.K - kk);    // remaining K extent (OOB -> 0)
      unsigned td1 = (unsigned)(p.Nc - col0); // remaining N rows  (OOB -> 0)
      v4u g0;
      g0[0] = 1u;                                             // count=1
      g0[1] = ldsoff;                                         // lds_addr
      g0[2] = (unsigned)(ga & 0xffffffffu);                   // global_addr lo
      g0[3] = (unsigned)((ga >> 32) & 0x01ffffffu) | (2u<<30);// addr hi | type=2
      v8i g1;
      g1[0] = (int)((1u<<16) | (1u<<20) | (4u<<22) | (7u<<25)); // 2B elems, pad
      g1[1] = (int)((td0 & 0xffffu) << 16);
      g1[2] = (int)(((td0 >> 16) & 0xffffu) | ((td1 & 0xffffu) << 16));
      g1[3] = (int)(((td1 >> 16) & 0xffffu) | (64u << 16));     // tile_dim0=64
      g1[4] = 64;                                               // tile_dim1=64
      g1[5] = (int)(unsigned)p.K;                               // dim0 stride
      g1[6] = 0; g1[7] = 0;
      v4i g2 = {0,0,0,0}, g3 = {0,0,0,0};
#if __clang_major__ >= 23
      v8i g4 = {0,0,0,0,0,0,0,0};
      __builtin_amdgcn_tensor_load_to_lds(g0, g1, g2, g3, g4, 0);
#else
      __builtin_amdgcn_tensor_load_to_lds(g0, g1, g2, g3, 0);
#endif
      __builtin_amdgcn_s_wait_tensorcnt(0);
    }
#else
    for (int t = 0; t < 16; ++t) {
      int idx = tid + t*256;
      int n = idx >> 6, c = idx & 63;
      int gn = col0 + n, gk = kk + c;
      _Float16 v = (_Float16)0.f;
      if (gn < p.Nc && gk < p.K) v = Bh[(long)gn*p.K + gk];
      Bs[n][c] = v;
    }
#endif
    __syncthreads();
    #pragma unroll
    for (int s = 0; s < 2; ++s) {
      const int kb = s*32 + sel*8;
      v8h alo = *(const v8h*)&As[wm + hl][kb];
      v8h ahi = *(const v8h*)&As[wm + hl][kb + 16];
      v16h av;
      #pragma unroll
      for (int e = 0; e < 8; ++e) { av[e] = alo[e]; av[e+8] = ahi[e]; }
      v8h b0lo = *(const v8h*)&Bs[wn + hl][kb];
      v8h b0hi = *(const v8h*)&Bs[wn + hl][kb + 16];
      v8h b1lo = *(const v8h*)&Bs[wn + 16 + hl][kb];
      v8h b1hi = *(const v8h*)&Bs[wn + 16 + hl][kb + 16];
      v16h bv0, bv1;
      #pragma unroll
      for (int e = 0; e < 8; ++e) {
        bv0[e] = b0lo[e]; bv0[e+8] = b0hi[e];
        bv1[e] = b1lo[e]; bv1[e+8] = b1hi[e];
      }
      acc[0] = __builtin_amdgcn_wmma_f32_16x16x32_f16(false, av, false, bv0,
                                                      (short)0, acc[0], false, false);
      acc[1] = __builtin_amdgcn_wmma_f32_16x16x32_f16(false, av, false, bv1,
                                                      (short)0, acc[1], false, false);
    }
    __syncthreads();
  }
  #pragma unroll
  for (int f = 0; f < 2; ++f) {
    const int ccol = col0 + wn + f*16 + hl;
    if (ccol < p.Nc) {
      #pragma unroll
      for (int j = 0; j < 8; ++j) {
        int r = row0 + wm + sel*8 + j;
        if (r < p.Mr) D[(long)r*p.ldc + ccol] = acc[f][j] + bias[ccol];
      }
    }
  }
}

// per-layer weight convert+transpose: W [m][K][Nc] f32 -> Wh [m][Nc][K] f16
__global__ void convert_wT_kernel(const float* __restrict__ W, _Float16* __restrict__ Wh,
                                  int K, int Nc, long Wo) {
  long total = (long)MM * Nc * K;
  long i = (long)blockIdx.x * 256 + threadIdx.x;
  if (i >= total) return;
  int  k = (int)(i % K);
  long t = i / K;
  int  n = (int)(t % Nc);
  int  m = (int)(t / Nc);
  Wh[i] = (_Float16)W[(long)m*Wo + (long)k*Nc + n];
}

// ---------------- small support kernels ----------------
__global__ void fill1_kernel(float* p, long n) {
  long i = (long)blockIdx.x * 256 + threadIdx.x;
  if (i < n) p[i] = 1.f;
}

__global__ void patch_gather_kernel(const float* __restrict__ x, float* __restrict__ pt) {
  long i = (long)blockIdx.x * 256 + threadIdx.x;
  long total = (long)BB * 196 * CC;
  if (i >= total) return;
  int j = (int)(i % CC);
  int g = (int)((i / CC) % 196);
  int b = (int)(i / ((long)CC * 196));
  int c = j >> 8, rem = j & 255, py = rem >> 4, px = rem & 15;
  int gy = g / GRD, gx = g % GRD;
  pt[i] = x[(((long)b*3 + c)*IMGSZ + gy*PP + py)*IMGSZ + gx*PP + px];
}

__global__ void assemble_kernel(const float* __restrict__ xt, const float* __restrict__ cls,
                                const float* __restrict__ pos, float* __restrict__ xc) {
  long i = (long)blockIdx.x * 256 + threadIdx.x;
  long total = (long)BB * NP0 * CC;
  if (i >= total) return;
  int c = (int)(i % CC);
  int n = (int)((i / CC) % NP0);
  int b = (int)(i / ((long)CC * NP0));
  float v = (n == 0) ? cls[c] : xt[((long)b*196 + (n-1))*CC + c];
  xc[i] = v + pos[(long)n*CC + c];
  (void)b;
}

__global__ __launch_bounds__(256) void ln_kernel(const float* __restrict__ x,
                                                 float* __restrict__ y) {
  long row = blockIdx.x;
  const float* xr = x + row * CC;
  float s = 0.f, ss = 0.f;
  for (int c = threadIdx.x; c < CC; c += 256) { float v = xr[c]; s += v; ss += v*v; }
  __shared__ float r1[256], r2[256];
  r1[threadIdx.x] = s; r2[threadIdx.x] = ss; __syncthreads();
  for (int off = 128; off; off >>= 1) {
    if (threadIdx.x < off) { r1[threadIdx.x] += r1[threadIdx.x+off]; r2[threadIdx.x] += r2[threadIdx.x+off]; }
    __syncthreads();
  }
  float mu  = r1[0] * (1.f / CC);
  float var = r2[0] * (1.f / CC) - mu * mu;
  float inv = rsqrtf(var + LNEPS);
  for (int c = threadIdx.x; c < CC; c += 256) y[row*CC + c] = (xr[c] - mu) * inv;
}

__global__ __launch_bounds__(256) void router_kernel(const float* __restrict__ xcur,
                                                     const float* __restrict__ rt,
                                                     float* __restrict__ rs, int N) {
  int b = blockIdx.x;
  const float* xr = xcur + (long)b * N * CC;
  float acc[MM];
  #pragma unroll
  for (int m = 0; m < MM; ++m) acc[m] = 0.f;
  for (int c = threadIdx.x; c < CC; c += 256) {
    float v = xr[c];
    #pragma unroll
    for (int m = 0; m < MM; ++m) acc[m] += v * rt[(long)c*MM + m];
  }
  __shared__ float red[256];
  __shared__ float outv[MM];
  for (int m = 0; m < MM; ++m) {
    red[threadIdx.x] = acc[m]; __syncthreads();
    for (int off = 128; off; off >>= 1) {
      if (threadIdx.x < off) red[threadIdx.x] += red[threadIdx.x+off];
      __syncthreads();
    }
    if (threadIdx.x == 0) outv[m] = red[0];
    __syncthreads();
  }
  if (threadIdx.x == 0) {
    float mx = outv[0];
    for (int m = 1; m < MM; ++m) mx = fmaxf(mx, outv[m]);
    float sum = 0.f;
    float e[MM];
    for (int m = 0; m < MM; ++m) { e[m] = expf(outv[m] - mx); sum += e[m]; }
    for (int m = 0; m < MM; ++m) rs[b*MM + m] = e[m] / sum;
  }
}

__global__ __launch_bounds__(256) void lora_down_kernel(const float* __restrict__ xn,
    const float* __restrict__ w1, const float* __restrict__ b1,
    const float* __restrict__ la, float* __restrict__ lo, int N) {
  long t = blockIdx.x;                       // over M*B*N tokens
  int  m  = (int)(t / ((long)BB * N));
  long bn = t % ((long)BB * N);
  const float* xr = xn + bn * CC;
  const float* aw = w1 + (long)m * LL * CC;
  const float* ab = b1 + (long)m * LL * CC;
  const float* A  = la + (long)m * LL * CC * LRA;
  float acc[LRA];
  #pragma unroll
  for (int r = 0; r < LRA; ++r) acc[r] = 0.f;
  for (int c = threadIdx.x; c < CC; c += 256) {
    float v = xr[c] * aw[c] + ab[c];
    #pragma unroll
    for (int r = 0; r < LRA; ++r) acc[r] += v * A[(long)c*LRA + r];
  }
  __shared__ float red[256];
  for (int r = 0; r < LRA; ++r) {
    red[threadIdx.x] = acc[r]; __syncthreads();
    for (int off = 128; off; off >>= 1) {
      if (threadIdx.x < off) red[threadIdx.x] += red[threadIdx.x+off];
      __syncthreads();
    }
    if (threadIdx.x == 0) lo[t*LRA + r] = red[0];
    __syncthreads();
  }
}

__global__ void lora_up_kernel(const float* __restrict__ lo, const float* __restrict__ lb,
                               float* __restrict__ qkv, int N) {
  long total = (long)MM * BB * N * C3;
  long i = (long)blockIdx.x * 256 + threadIdx.x;
  if (i >= total) return;
  int  d = (int)(i % C3);
  long t = i / C3;
  int  m = (int)(t / ((long)BB * N));
  const float* l = lo + t * LRA;
  const float* w = lb + (long)m * LL * LRA * C3 + d;
  float s = 0.f;
  #pragma unroll
  for (int r = 0; r < LRA; ++r) s += l[r] * w[(long)r * C3];
  qkv[i] += s;
}

__global__ __launch_bounds__(256) void attn_softmax_kernel(float* __restrict__ S,
                                                           const float* __restrict__ sz, int N) {
  long row = blockIdx.x;                     // (((m*B+b)*H + h)*N + n)
  long mb  = row / ((long)HH * N);
  float* sr = S + row * N;
  const float* s0 = sz + mb * N;
  __shared__ float red[256];
  float mx = -3.4e38f;
  for (int k = threadIdx.x; k < N; k += 256) mx = fmaxf(mx, sr[k] + logf(s0[k]));
  red[threadIdx.x] = mx; __syncthreads();
  for (int off = 128; off; off >>= 1) {
    if (threadIdx.x < off) red[threadIdx.x] = fmaxf(red[threadIdx.x], red[threadIdx.x+off]);
    __syncthreads();
  }
  mx = red[0]; __syncthreads();
  float sum = 0.f;
  for (int k = threadIdx.x; k < N; k += 256) {
    float e = expf(sr[k] + logf(s0[k]) - mx);
    sr[k] = e; sum += e;
  }
  red[threadIdx.x] = sum; __syncthreads();
  for (int off = 128; off; off >>= 1) {
    if (threadIdx.x < off) red[threadIdx.x] += red[threadIdx.x+off];
    __syncthreads();
  }
  float inv = 1.f / red[0];
  for (int k = threadIdx.x; k < N; k += 256) sr[k] *= inv;
}

__global__ void metric_kernel(const float* __restrict__ qkv, float* __restrict__ met, int N) {
  long total = (long)MM * BB * N * HD;
  long i = (long)blockIdx.x * 256 + threadIdx.x;
  if (i >= total) return;
  int  d   = (int)(i % HD);
  long mbn = i / HD;
  const float* kp = qkv + mbn * C3 + CC + d;
  float s = 0.f;
  #pragma unroll
  for (int h = 0; h < HH; ++h) s += kp[h * HD];
  met[i] = s * (1.f / HH);
}

// ToMe merge: one block per (m,b). N <= 197 -> na <= 99.
__global__ __launch_bounds__(256) void merge_kernel(const float* __restrict__ metric,
    const float* __restrict__ xproc, const float* __restrict__ szi_g,
    float* __restrict__ xout, float* __restrict__ szo_g, int N) {
  int mb = blockIdx.x;
  int na = (N + 1) >> 1, nb = N >> 1, nunm = na - RR;
  const float* met = metric + (long)mb * N * HD;
  const float* xp  = xproc  + (long)mb * N * CC;
  const float* szi = szi_g  + (long)mb * N;
  float* xo  = xout  + (long)mb * (N - RR) * CC;
  float* szo = szo_g + (long)mb * (N - RR);
  __shared__ float inva[128], invb[128], nmax[128];
  __shared__ int nidx[128], edge[128], unm[128], srcI[RR], dstI[RR];
  int tid = threadIdx.x, lane = tid & 31, w = tid >> 5;
  // inverse norms of metric rows
  for (int r = tid; r < N; r += 256) {
    const float* v = met + (long)r * HD;
    float s = 0.f;
    for (int d = 0; d < HD; ++d) s += v[d] * v[d];
    float inv = rsqrtf(fmaxf(s, 1e-30f));
    if (r & 1) invb[r >> 1] = inv; else inva[r >> 1] = inv;
  }
  __syncthreads();
  // per even-row max cosine similarity vs odd rows (one warp per row)
  for (int ia = w; ia < na; ia += 8) {
    float best = -3.4e38f; int bidx = 0;
    const float* av = met + (long)(2*ia) * HD;
    float as = inva[ia];
    for (int ib = lane; ib < nb; ib += 32) {
      const float* bv = met + (long)(2*ib + 1) * HD;
      float dot = 0.f;
      for (int d = 0; d < HD; ++d) dot += av[d] * bv[d];
      dot *= as * invb[ib];
      if (dot > best) { best = dot; bidx = ib; }
    }
    for (int off = 16; off; off >>= 1) {
      float ob = __shfl_down(best, off, 32);
      int   oi = __shfl_down(bidx, off, 32);
      if (ob > best || (ob == best && oi < bidx)) { best = ob; bidx = oi; }
    }
    if (lane == 0) {
      nmax[ia] = (ia == 0) ? -3.4e38f : best;   // protect CLS row
      nidx[ia] = (ia == 0) ? 0 : bidx;
    }
  }
  __syncthreads();
  // stable descending argsort of nmax via O(n^2) ranking
  for (int e = tid; e < na; e += 256) {
    float ki = nmax[e]; int rank = 0;
    for (int j = 0; j < na; ++j) {
      float kj = nmax[j];
      if (kj > ki || (kj == ki && j < e)) rank++;
    }
    edge[rank] = e;
  }
  __syncthreads();
  if (tid < RR) { srcI[tid] = edge[tid]; dstI[tid] = nidx[edge[tid]]; }
  __syncthreads();
  // ascending sort of unmerged indices
  for (int e = tid; e < nunm; e += 256) {
    int v = edge[RR + e]; int rank = 0;
    for (int j = 0; j < nunm; ++j) if (edge[RR + j] < v) rank++;
    unm[rank] = v;
  }
  __syncthreads();
  // unmerged rows: weighted-avg of a single source == passthrough; size carried
  for (int r = 0; r < nunm; ++r) {
    int t = 2 * unm[r];
    for (int c = tid; c < CC; c += 256) xo[(long)r*CC + c] = xp[(long)t*CC + c];
    if (tid == 0) szo[r] = szi[t];
  }
  // destination rows: size-weighted scatter-add then renormalize
  for (int j2 = 0; j2 < nb; ++j2) {
    int t = 2*j2 + 1;
    int nsrc = 0, slist[RR];
    float stot = szi[t];
    for (int r = 0; r < RR; ++r) if (dstI[r] == j2) slist[nsrc++] = 2 * srcI[r];
    for (int r = 0; r < nsrc; ++r) stot += szi[slist[r]];
    float invs = 1.f / stot;
    for (int c = tid; c < CC; c += 256) {
      float acc = xp[(long)t*CC + c] * szi[t];
      for (int r = 0; r < nsrc; ++r) acc += xp[(long)slist[r]*CC + c] * szi[slist[r]];
      xo[(long)(nunm + j2)*CC + c] = acc * invs;
    }
    if (tid == 0) szo[nunm + j2] = stot;
  }
}

__global__ void combine_kernel(const float* __restrict__ xm, const float* __restrict__ rs,
                               float* __restrict__ mg, int nout) {
  long total = (long)BB * nout * CC;
  long i = (long)blockIdx.x * 256 + threadIdx.x;
  if (i >= total) return;
  int  c = (int)(i % CC);
  long t = i / CC;
  int  b = (int)(t / nout);
  int  n = (int)(t % nout);
  float s = 0.f;
  #pragma unroll
  for (int m = 0; m < MM; ++m)
    s += xm[(((long)m*BB + b)*nout + n)*CC + c] * rs[b*MM + m];
  mg[i] = s;
}

// ---------------- host side ----------------
static inline void gemm(hipStream_t st, int Mr, int Nc, int K, int nb, int dv,
    const float* A, int lda, long Ai, long Ao,
    const float* Bm, int ldb, int tB, long Bi, long Bo,
    const float* aw, const float* ab, long AWi, long AWo,
    const float* bias, long BsI, long BsO,
    const float* a1, int ld1, long A1i, long A1o,
    const float* a2, int ld2, long A2i, long A2o,
    float* D, int ldc, long Di, long Do, float alpha, int act) {
  GemmP p{Mr, Nc, K, dv, A, lda, Ai, Ao, Bm, ldb, tB, Bi, Bo,
          aw, ab, AWi, AWo, bias, BsI, BsO,
          a1, ld1, A1i, A1o, a2, ld2, A2i, A2o,
          D, ldc, Di, Do, alpha, act};
  dim3 g((Mr + TM - 1)/TM, (Nc + TN - 1)/TN, nb);
  gemm_wmma_kernel<<<g, dim3(256), 0, st>>>(p);
}

extern "C" void kernel_launch(void* const* d_in, const int* in_sizes, int n_in,
                              void* d_out, int out_size, void* d_ws, size_t ws_size,
                              hipStream_t stream) {
  (void)in_sizes; (void)n_in; (void)out_size; (void)ws_size;
  const float* x       = (const float*)d_in[0];
  const float* routers = (const float*)d_in[1];
  const float* patch_w = (const float*)d_in[2];
  const float* patch_b = (const float*)d_in[3];
  const float* cls_tok = (const float*)d_in[4];
  const float* pos_emb = (const float*)d_in[5];
  const float* norm1_w = (const float*)d_in[6];
  const float* norm1_b = (const float*)d_in[7];
  const float* qkv_w   = (const float*)d_in[8];
  const float* qkv_b   = (const float*)d_in[9];
  const float* lora_a  = (const float*)d_in[10];
  const float* lora_bw = (const float*)d_in[11];
  const float* proj_w  = (const float*)d_in[12];
  const float* proj_b  = (const float*)d_in[13];
  const float* norm2_w = (const float*)d_in[14];
  const float* norm2_b = (const float*)d_in[15];
  const float* fc1_w   = (const float*)d_in[16];
  const float* fc1_b   = (const float*)d_in[17];
  const float* fc2_w   = (const float*)d_in[18];
  const float* fc2_b   = (const float*)d_in[19];
  const float* ad_dw   = (const float*)d_in[20];
  const float* ad_db   = (const float*)d_in[21];
  const float* ad_uw   = (const float*)d_in[22];
  const float* ad_ub   = (const float*)d_in[23];
  const float* normf_w = (const float*)d_in[24];
  const float* normf_b = (const float*)d_in[25];
  const float* head_w  = (const float*)d_in[26];
  const float* head_b  = (const float*)d_in[27];
  float* out = (float*)d_out;

  char* base = (char*)d_ws; size_t off = 0;
  auto alloc = [&](long nelem) -> float* {
    float* p = (float*)(base + off);
    off += ((size_t)nelem * sizeof(float) + 255) & ~(size_t)255;
    return p;
  };
  float* patches = alloc((long)BB*196*CC);
  float* xt   = alloc((long)BB*196*CC);
  float* xcur = alloc((long)BB*NP0*CC);
  float* xn   = alloc((long)BB*NP0*CC);
  float* rsc  = alloc((long)BB*MM);
  float* qkv  = alloc((long)MM*BB*NP0*C3);
  float* lo   = alloc((long)MM*BB*NP0*LRA);
  float* S    = alloc((long)MM*BB*HH*NP0*NP0);
  float* att  = alloc((long)MM*BB*NP0*CC);
  float* xpr  = alloc((long)MM*BB*NP0*CC);
  float* met  = alloc((long)MM*BB*NP0*HD);
  float* szA  = alloc((long)MM*BB*NP0);
  float* szB  = alloc((long)MM*BB*NP0);
  float* xm   = alloc((long)MM*BB*NP0*CC);
  float* mg   = alloc((long)BB*NP0*CC);
  float* h4   = alloc((long)BB*NP0*C4);
  float* hb   = alloc((long)BB*NP0*CC);
  float* adh  = alloc((long)BB*NP0*ADHN);
  _Float16* qkvwh = (_Float16*)alloc((long)MM*C3*CC / 2);  // f16 [m][3C][C]

  auto g1 = [](long n) { return dim3((unsigned)((n + 255) / 256)); };

  // patch embed: gather -> GEMM -> cls/pos assemble
  patch_gather_kernel<<<g1((long)BB*196*CC), 256, 0, stream>>>(x, patches);
  gemm(stream, BB*196, CC, CC, 1, 1,
       patches, CC, 0, 0,
       patch_w, CC, 0, 0, 0,
       nullptr, nullptr, 0, 0,
       patch_b, 0, 0,
       nullptr, 0, 0, 0, nullptr, 0, 0, 0,
       xt, CC, 0, 0, 1.f, 0);
  assemble_kernel<<<g1((long)BB*NP0*CC), 256, 0, stream>>>(xt, cls_tok, pos_emb, xcur);
  fill1_kernel<<<g1((long)MM*BB*NP0), 256, 0, stream>>>(szA, (long)MM*BB*NP0);

  float* szCur = szA; float* szNext = szB;
  for (int i = 0; i < LL; ++i) {
    int N = NP0 - RR * i;
    int Nout = N - RR;
    long BN = (long)BB * N;
    long BNo = (long)BB * Nout;

    router_kernel<<<BB, 256, 0, stream>>>(xcur, routers + (long)i*CC*MM, rsc, N);
    ln_kernel<<<(unsigned)BN, 256, 0, stream>>>(xcur, xn);

    // QKV = (ln * n1w + n1b) @ qkv_w + qkv_b  (experts batched; TDM B staging)
    convert_wT_kernel<<<g1((long)MM*C3*CC), 256, 0, stream>>>(
        qkv_w + (long)i*CC*C3, qkvwh, CC, C3, (long)LL*CC*C3);
    {
      GemmHT hp{xn, CC, (int)BN,
                qkvwh, (long)C3*CC, C3, CC,
                norm1_w + (long)i*CC, norm1_b + (long)i*CC, (long)LL*CC,
                qkv_b + (long)i*C3, (long)LL*C3,
                qkv, C3, BN*C3};
      dim3 gq(((int)BN + TM - 1)/TM, (C3 + TN - 1)/TN, MM);
      gemm_wmma_tdm_kernel<<<gq, dim3(256), 0, stream>>>(hp);
    }
    // LoRA path (K=8: VALU is the right tool)
    lora_down_kernel<<<(unsigned)(MM*BN), 256, 0, stream>>>(xn,
        norm1_w + (long)i*CC, norm1_b + (long)i*CC,
        lora_a + (long)i*CC*LRA, lo, N);
    lora_up_kernel<<<g1((long)MM*BN*C3), 256, 0, stream>>>(lo, lora_bw + (long)i*LRA*C3, qkv, N);

    // scores = scale * q @ k^T, batched over (m*B+b, h)
    gemm(stream, N, N, HD, MM*BB*HH, HH,
         qkv, C3, HD, (long)N*C3,
         qkv + CC, C3, 1, HD, (long)N*C3,
         nullptr, nullptr, 0, 0,
         nullptr, 0, 0,
         nullptr, 0, 0, 0, nullptr, 0, 0, 0,
         S, N, (long)N*N, (long)HH*N*N, ATT_SCALE, 0);
    attn_softmax_kernel<<<(unsigned)((long)MM*BB*HH*N), 256, 0, stream>>>(S, szCur, N);
    // out = softmax @ v
    gemm(stream, N, HD, N, MM*BB*HH, HH,
         S, N, (long)N*N, (long)HH*N*N,
         qkv + 2*CC, C3, 0, HD, (long)N*C3,
         nullptr, nullptr, 0, 0,
         nullptr, 0, 0,
         nullptr, 0, 0, 0, nullptr, 0, 0, 0,
         att, CC, HD, (long)N*CC, 1.f, 0);

    metric_kernel<<<g1((long)MM*BN*HD), 256, 0, stream>>>(qkv, met, N);

    // xproc = xcur + out @ proj_w + proj_b
    gemm(stream, (int)BN, CC, CC, MM, 1,
         att, CC, 0, BN*CC,
         proj_w + (long)i*CC*CC, CC, 0, 0, (long)LL*CC*CC,
         nullptr, nullptr, 0, 0,
         proj_b + (long)i*CC, 0, (long)LL*CC,
         xcur, CC, 0, 0,
         nullptr, 0, 0, 0,
         xpr, CC, 0, BN*CC, 1.f, 0);

    merge_kernel<<<MM*BB, 256, 0, stream>>>(met, xpr, szCur, xm, szNext, N);
    combine_kernel<<<g1((long)BNo*CC), 256, 0, stream>>>(xm, rsc, mg, Nout);

    ln_kernel<<<(unsigned)BNo, 256, 0, stream>>>(mg, xn);
    // fc1 with fused norm2 affine + GELU
    gemm(stream, (int)BNo, C4, CC, 1, 1,
         xn, CC, 0, 0,
         fc1_w + (long)i*CC*C4, C4, 0, 0, 0,
         norm2_w + (long)i*CC, norm2_b + (long)i*CC, 0, 0,
         fc1_b + (long)i*C4, 0, 0,
         nullptr, 0, 0, 0, nullptr, 0, 0, 0,
         h4, C4, 0, 0, 1.f, 1);
    gemm(stream, (int)BNo, CC, C4, 1, 1,
         h4, C4, 0, 0,
         fc2_w + (long)i*C4*CC, CC, 0, 0, 0,
         nullptr, nullptr, 0, 0,
         fc2_b + (long)i*CC, 0, 0,
         nullptr, 0, 0, 0, nullptr, 0, 0, 0,
         hb, CC, 0, 0, 1.f, 0);
    // adapter down + ReLU
    gemm(stream, (int)BNo, ADHN, CC, 1, 1,
         hb, CC, 0, 0,
         ad_dw + (long)i*CC*ADHN, ADHN, 0, 0, 0,
         nullptr, nullptr, 0, 0,
         ad_db + (long)i*ADHN, 0, 0,
         nullptr, 0, 0, 0, nullptr, 0, 0, 0,
         adh, ADHN, 0, 0, 1.f, 2);
    // xcur_next = merged + h + relu(...) @ ad_uw + ad_ub
    gemm(stream, (int)BNo, CC, ADHN, 1, 1,
         adh, ADHN, 0, 0,
         ad_uw + (long)i*ADHN*CC, CC, 0, 0, 0,
         nullptr, nullptr, 0, 0,
         ad_ub + (long)i*CC, 0, 0,
         hb, CC, 0, 0,
         mg, CC, 0, 0,
         xcur, CC, 0, 0, 1.f, 0);

    float* tsw = szCur; szCur = szNext; szNext = tsw;
  }

  int Nf = NP0 - RR * LL;  // 101
  ln_kernel<<<(unsigned)((long)BB*Nf), 256, 0, stream>>>(xcur, xn);
  // logits[b, m*NC+k] = (ln(xcur)[b,0]*nfw+nfb) @ head_w[m] + head_b[m]
  gemm(stream, BB, NCLS, CC, MM, 1,
       xn, Nf*CC, 0, 0,
       head_w, NCLS, 0, 0, (long)CC*NCLS,
       normf_w, normf_b, 0, 0,
       head_b, 0, NCLS,
       nullptr, 0, 0, 0, nullptr, 0, 0, 0,
       out, MM*NCLS, 0, NCLS, 1.f, 0);
}
